// BinaryConv2d_49168785605334
// MI455X (gfx1250) — matrix operand order repaired and checked by reading, not compiled, not run on hardware
//
#include <hip/hip_runtime.h>

typedef __attribute__((ext_vector_type(16))) __bf16         v16bf;
typedef __attribute__((ext_vector_type(8)))  float          v8f;
typedef __attribute__((ext_vector_type(8)))  unsigned int   v8u;
typedef __attribute__((ext_vector_type(2)))  unsigned short v2us;

#define HW      112
#define IMG     (HW * HW)          // 12544 pixels per image-channel
#define C_IN    64
#define K_OUT   64
#define NIMG    32
#define KP      288                // 576 reduction elems -> 288 packed bf16 pairs
#define ROWSTR  292                // padded dword stride per out-channel row in LDS
#define LDS_DW  (K_OUT * ROWSTR)   // 18688 dwords = 74752 bytes
#define NPIX    256                // pixels per block (8 waves x 32 pixels)

// float x2 -> packed bf16 pair (RNE-ish); used only for one-time weight staging
__device__ __forceinline__ unsigned int pack_bf16(float a, float b) {
    unsigned int ua = __builtin_bit_cast(unsigned int, a) + 0x8000u;
    unsigned int ub = __builtin_bit_cast(unsigned int, b) + 0x8000u;
    return (ua >> 16) | (ub & 0xFFFF0000u);
}

extern "C" __global__ __launch_bounds__(256)
void binconv3x3_wmma_bf16(const float* __restrict__ in,
                          const float* __restrict__ wgt,
                          const float* __restrict__ alpha,
                          float* __restrict__ out)
{
    extern __shared__ __align__(16) unsigned int ldsA[];   // LDS_DW dwords

    const int tid  = threadIdx.x;
    const int lane = tid & 31;
    const int wave = tid >> 5;
    const int lrow = lane & 15;    // row within 16 / pixel within tile
    const int g    = lane >> 4;    // lane-group: K half selector

    // ------------------------------------------------------------------
    // Stage binarized weights into LDS as bf16 pairs (exact: +/- alpha).
    // Reduction index kidx = rs*64 + c  (rs = r*3+s); pair kp = kidx/2.
    // ------------------------------------------------------------------
    for (int i = tid; i < K_OUT * KP; i += 256) {
        const int row = i / KP;                 // output channel
        const int kp  = i - row * KP;
        const int k0  = kp * 2;
        const int rs  = k0 >> 6;                // tap index 0..8
        const int c   = k0 & 63;                // input channel (even)
        const float a  = alpha[row];
        const float w0 = wgt[(row * C_IN + c    ) * 9 + rs];
        const float w1 = wgt[(row * C_IN + c + 1) * 9 + rs];
        const float b0 = (w0 >= 0.0f) ? a : -a;
        const float b1 = (w1 >= 0.0f) ? a : -a;
        ldsA[row * ROWSTR + kp] = pack_bf16(b0, b1);
    }
    __syncthreads();

    // ------------------------------------------------------------------
    // Per-wave tile: all 64 out channels x 32 pixels (2 N-tiles of 16).
    // ------------------------------------------------------------------
    const int pixBase = blockIdx.x * NPIX + wave * 32;
    int h[2], w[2], inBase[2], outBase[2];
    #pragma unroll
    for (int nt = 0; nt < 2; ++nt) {
        const int p   = pixBase + nt * 16 + lrow;
        const int n   = p / IMG;
        const int rem = p - n * IMG;
        h[nt] = rem / HW;
        w[nt] = rem - h[nt] * HW;
        inBase[nt]  = n * (C_IN  * IMG) + rem;
        outBase[nt] = n * (K_OUT * IMG) + rem;
    }

    const v8f zero = {0.f, 0.f, 0.f, 0.f, 0.f, 0.f, 0.f, 0.f};
    v8f acc[4][2];
    #pragma unroll
    for (int m = 0; m < 4; ++m)
        #pragma unroll
        for (int nt = 0; nt < 2; ++nt)
            acc[m][nt] = zero;

    // ------------------------------------------------------------------
    // K loop: 9 taps x 2 channel-halves (K=32 each). Tap setup (dr, dc,
    // validity mask, padding offset) is shared by both halves.
    // ------------------------------------------------------------------
    #pragma unroll 1
    for (int rs = 0; rs < 9; ++rs) {
        const int dr = rs / 3 - 1;
        const int dc = rs - (rs / 3) * 3 - 1;

        unsigned int msk[2];
        const unsigned short* sp[2];
        #pragma unroll
        for (int nt = 0; nt < 2; ++nt) {
            const int  hh    = h[nt] + dr;
            const int  ww    = w[nt] + dc;
            const bool valid = (hh >= 0) && (hh < HW) && (ww >= 0) && (ww < HW);
            msk[nt] = valid ? 0xFFFFFFFFu : 0u;
            const int off = valid ? (dr * HW + dc) : 0;      // safe in-bounds addr
            sp[nt] = reinterpret_cast<const unsigned short*>(
                         in + inBase[nt] + off + g * 16 * IMG);
        }

        #pragma unroll
        for (int half = 0; half < 2; ++half) {
            const int kk = rs * 2 + half;                    // K chunk index

            // B fragments: K=32 x N=16 bf16. Lane group g holds K half,
            // VGPR v holds K pair (g*16+2v, +1); N = lane%16.
            // bf16 truncation == high 16 bits of the f32, so load just the
            // upper 2 bytes of each float into the lo/hi half of the pair.
            v16bf B[2];
            #pragma unroll
            for (int nt = 0; nt < 2; ++nt) {
                const unsigned short* s =
                    sp[nt] + (half * 32) * (2 * IMG);        // channel half
                if (nt == 0 && half == 0)
                    __builtin_prefetch(s + 32 * (2 * IMG), 0, 0);
                v8u bw;
                #pragma unroll
                for (int v = 0; v < 8; ++v) {
                    v2us pr;
                    pr.x = s[(2 * v    ) * (2 * IMG) + 1];   // K even -> lo half
                    pr.y = s[(2 * v + 1) * (2 * IMG) + 1];   // K odd  -> hi half
                    bw[v] = __builtin_bit_cast(unsigned int, pr) & msk[nt];
                }
                B[nt] = __builtin_bit_cast(v16bf, bw);
            }

            // A fragments from LDS: M=16 x K=32 bf16 per tile.
            // Lane holds row M=lane%16; VGPR v -> pair kk*16 + g*8 + v     (v<8)
            #pragma unroll
            for (int m = 0; m < 4; ++m) {
                const int row = m * 16 + lrow;
                const unsigned int* base = &ldsA[row * ROWSTR + kk * 16 + g * 8];
                const uint4 lo = *reinterpret_cast<const uint4*>(base);
                const uint4 hi = *reinterpret_cast<const uint4*>(base + 4);
                v8u aw;
                aw[0] = lo.x; aw[1] = lo.y; aw[2] = lo.z; aw[3] = lo.w;
                aw[4] = hi.x; aw[5] = hi.y; aw[6] = hi.z; aw[7] = hi.w;
                const v16bf A = __builtin_bit_cast(v16bf, aw);
                #pragma unroll
                for (int nt = 0; nt < 2; ++nt) {
                    acc[m][nt] = __builtin_amdgcn_wmma_f32_16x16x32_bf16(
                        false, A, false, B[nt], (short)0, acc[m][nt], false, false);
                }
            }
        }
    }

    // ------------------------------------------------------------------
    // Writeback: C/D layout -> lanes 0-15: M=j, lanes 16-31: M=8+j; N=lane%16.
    // ------------------------------------------------------------------
    #pragma unroll
    for (int m = 0; m < 4; ++m) {
        #pragma unroll
        for (int nt = 0; nt < 2; ++nt) {
            #pragma unroll
            for (int j = 0; j < 8; ++j) {
                const int kout = m * 16 + g * 8 + j;
                out[outBase[nt] + kout * IMG] = acc[m][nt][j];
            }
        }
    }
}

extern "C" void kernel_launch(void* const* d_in, const int* in_sizes, int n_in,
                              void* d_out, int out_size, void* d_ws, size_t ws_size,
                              hipStream_t stream) {
    (void)in_sizes; (void)n_in; (void)out_size; (void)d_ws; (void)ws_size;
    const float* in  = (const float*)d_in[0];   // (32,64,112,112) f32
    const float* wg  = (const float*)d_in[1];   // (64,64,3,3)     f32
    const float* al  = (const float*)d_in[2];   // (64,1,1,1)      f32
    float* out       = (float*)d_out;           // (32,64,112,112) f32

    const int totalPix = NIMG * IMG;            // 401408, divisible by 256
    dim3 grid(totalPix / NPIX);                 // 1568 blocks
    dim3 block(256);                            // 8 wave32
    const size_t shmem = (size_t)LDS_DW * sizeof(unsigned int);  // 74752 B
    hipLaunchKernelGGL(binconv3x3_wmma_bf16, grid, block, shmem, stream,
                       in, wg, al, out);
}
